// SparseAttention_88304527606385
// MI455X (gfx1250) — matrix operand order repaired and checked by reading, not compiled, not run on hardware
//
#include <hip/hip_runtime.h>
#include <hip/hip_bf16.h>

// ---------------------------------------------------------------------------
// Sparse attention block for MI455X (gfx1250, wave32, WMMA bf16 path,
// async global->LDS staging). B=4, S=4096, D=512, top-k k = 1228.
// ---------------------------------------------------------------------------

typedef unsigned short bf16_t;
typedef __attribute__((ext_vector_type(16))) __bf16 v16bf;
typedef __attribute__((ext_vector_type(8)))  float  v8f;

#define D_MODEL 512
#define SEQ     4096
#define BATCH   4
#define MROWS   (BATCH * SEQ)      // 16384
#define N_QKV   (3 * D_MODEL)      // 1536
#define TOPK    1228
#define LN_EPS  1e-5f

// GEMM tiling: block = 256 threads = 8 waves (4 in M x 2 in N).
// Block tile 128x64, wave tile 32x32 (4 WMMA accumulators), K-step 32.
#define BM 128
#define BN 64
#define BK 32
#define LDP 40   // LDS row pitch in bf16 (80 B: uint4-aligned, bank-conflict-free)

__device__ __forceinline__ bf16_t f2bf(float f) {
    unsigned int u = __float_as_uint(f);
    unsigned int r = u + 0x7FFFu + ((u >> 16) & 1u);   // round-to-nearest-even
    return (bf16_t)(r >> 16);
}

__device__ __forceinline__ unsigned int mapf(float f) {
    // order-preserving f32 -> u32 (larger float => larger uint)
    unsigned int u = __float_as_uint(f);
    return (u & 0x80000000u) ? ~u : (u | 0x80000000u);
}

// LDS byte offset of a generic pointer to __shared__ data: the generic
// address's low 32 bits are the LDS allocation offset (ISA: LDS aperture
// mapping uses addr[31:0]).
__device__ __forceinline__ unsigned lds_off(const void* p) {
    return (unsigned)(size_t)p;
}

// Async DMA: 16 bytes per lane, global -> LDS, tracked by ASYNCcnt.
__device__ __forceinline__ void async_ld_b128(unsigned ldsOff, const void* g) {
    asm volatile("global_load_async_to_lds_b128 %0, %1, off"
                 :: "v"(ldsOff), "v"((unsigned long long)(size_t)g)
                 : "memory");
}

__device__ __forceinline__ void wait_async0() {
    asm volatile("s_wait_asynccnt 0x0" ::: "memory");
}

// ---------------------------------------------------------------------------
// 1) LayerNorm: one block (256 thr) per row of 512; outputs f32 + bf16 copies.
// ---------------------------------------------------------------------------
__global__ void __launch_bounds__(256)
k_layernorm(const float* __restrict__ x, const float* __restrict__ g,
            const float* __restrict__ b, float* __restrict__ outF,
            bf16_t* __restrict__ outH) {
    __shared__ float red[256];
    const int row = blockIdx.x;
    const int tid = threadIdx.x;
    const float* xr = x + (size_t)row * D_MODEL;

    float v0 = xr[tid];
    float v1 = xr[tid + 256];

    red[tid] = v0 + v1;
    __syncthreads();
    for (int o = 128; o > 0; o >>= 1) {
        if (tid < o) red[tid] += red[tid + o];
        __syncthreads();
    }
    float mu = red[0] * (1.0f / D_MODEL);
    __syncthreads();

    float d0 = v0 - mu, d1 = v1 - mu;
    red[tid] = d0 * d0 + d1 * d1;
    __syncthreads();
    for (int o = 128; o > 0; o >>= 1) {
        if (tid < o) red[tid] += red[tid + o];
        __syncthreads();
    }
    float rstd = rsqrtf(red[0] * (1.0f / D_MODEL) + LN_EPS);

    float y0 = d0 * rstd * g[tid]       + b[tid];
    float y1 = d1 * rstd * g[tid + 256] + b[tid + 256];
    size_t base = (size_t)row * D_MODEL;
    outF[base + tid]       = y0;
    outF[base + tid + 256] = y1;
    outH[base + tid]       = f2bf(y0);
    outH[base + tid + 256] = f2bf(y1);
}

// ---------------------------------------------------------------------------
// 2) Transpose + convert f32 weight [R,C] -> bf16 [C,R] (for NT-form GEMM).
//    grid = (C/32, R/32), block = (32,8).
// ---------------------------------------------------------------------------
__global__ void __launch_bounds__(256)
k_transpose_cvt(const float* __restrict__ W, bf16_t* __restrict__ WT,
                int R, int C) {
    __shared__ float tile[32][33];
    const int c0 = blockIdx.x * 32;
    const int r0 = blockIdx.y * 32;
    for (int i = threadIdx.y; i < 32; i += 8)
        tile[i][threadIdx.x] = W[(size_t)(r0 + i) * C + c0 + threadIdx.x];
    __syncthreads();
    for (int i = threadIdx.y; i < 32; i += 8)
        WT[(size_t)(c0 + i) * R + r0 + threadIdx.x] = f2bf(tile[threadIdx.x][i]);
}

// Transpose bf16 [R,C] (row stride srcStride) -> [C,R].
__global__ void __launch_bounds__(256)
k_transpose_bf16(const bf16_t* __restrict__ src, int srcStride,
                 bf16_t* __restrict__ dst, int R, int C) {
    __shared__ bf16_t tile[32][33];
    const int c0 = blockIdx.x * 32;
    const int r0 = blockIdx.y * 32;
    for (int i = threadIdx.y; i < 32; i += 8)
        tile[i][threadIdx.x] = src[(size_t)(r0 + i) * srcStride + c0 + threadIdx.x];
    __syncthreads();
    for (int i = threadIdx.y; i < 32; i += 8)
        dst[(size_t)(c0 + i) * R + r0 + threadIdx.x] = tile[threadIdx.x][i];
}

// ---------------------------------------------------------------------------
// 3) GEMM (NT form): C[M,N] = scale * A[M,K] * B[N,K]^T + bias (+ resid)
//    Double-buffered LDS fed by GLOBAL_LOAD_ASYNC_TO_LDS_B128 (ASYNCcnt),
//    one barrier per K-step, 4 WMMA per wave per K-step.
//    Requires M%128==0, N%64==0, K%32==0 (true at all call sites).
// ---------------------------------------------------------------------------
union FragAB { v16bf v; uint4 q[2]; };

__global__ void __launch_bounds__(256)
k_gemm_nt(const bf16_t* __restrict__ A, int lda,
          const bf16_t* __restrict__ Bm, int ldb,
          int M, int N, int K, float scale,
          const float* __restrict__ bias,
          const float* __restrict__ resid,
          float* __restrict__ outF,
          bf16_t* __restrict__ outH, int ldo) {
    __shared__ bf16_t As[2][BM * LDP];   // 10 KB per buffer
    __shared__ bf16_t Bs[2][BN * LDP];   // 5 KB per buffer

    const int tid  = threadIdx.x;
    const int wave = tid >> 5;          // wave32: 8 waves / block
    const int lane = tid & 31;
    const int l16  = lane & 15;
    const int half = lane >> 4;
    const int wm   = wave >> 1;         // 0..3 : M position
    const int wn   = wave & 1;          // 0..1 : N position

    const int mBase = blockIdx.y * BM;
    const int nBase = blockIdx.x * BN;

    // --- staging: A tile 128x32 (512 x b128, 2/thread), B tile 64x32 (1/thread)
    const int sRow  = tid >> 2;         // 0..63
    const int sColQ = tid & 3;          // b128 column within the 32-wide tile
    const bf16_t* aSrc0 = A  + (size_t)(mBase + sRow) * lda + sColQ * 8;
    const bf16_t* aSrc1 = aSrc0 + (size_t)64 * lda;
    const bf16_t* bSrc  = Bm + (size_t)(nBase + sRow) * ldb + sColQ * 8;
    const unsigned aOff0 = lds_off(&As[0][0] + sRow * LDP + sColQ * 8);
    const unsigned aOff1 = aOff0 + 64 * LDP * 2;
    const unsigned bOff  = lds_off(&Bs[0][0] + sRow * LDP + sColQ * 8);
    const unsigned aBufBytes = BM * LDP * 2;
    const unsigned bBufBytes = BN * LDP * 2;
    const int bufStrideA = BM * LDP;
    const int bufStrideB = BN * LDP;

    v8f zero = {};
    v8f acc[2][2];
    acc[0][0] = zero; acc[0][1] = zero; acc[1][0] = zero; acc[1][1] = zero;

    // prologue: DMA tile k0=0 -> buffer 0 (no VGPR round-trip)
    async_ld_b128(aOff0, aSrc0);
    async_ld_b128(aOff1, aSrc1);
    async_ld_b128(bOff,  bSrc);

    int buf = 0;
    for (int k0 = 0; k0 < K; k0 += BK) {
        wait_async0();                         // my DMA into `buf` landed
        __syncthreads();                       // everyone's DMA landed
        const int kn = k0 + BK;
        if (kn < K) {                          // DMA next tile -> other buffer
            const unsigned o = (unsigned)(buf ^ 1);
            async_ld_b128(aOff0 + o * aBufBytes, aSrc0 + kn);
            async_ld_b128(aOff1 + o * aBufBytes, aSrc1 + kn);
            async_ld_b128(bOff  + o * bBufBytes, bSrc + kn);
            if (kn + BK < K)
                __builtin_prefetch(aSrc0 + kn + BK, 0, 1);  // global_prefetch_b8
        }

        // fragments from LDS (conflict-free b128 reads)
        const bf16_t* aBuf = &As[0][0] + buf * bufStrideA;
        const bf16_t* bBuf = &Bs[0][0] + buf * bufStrideB;
        FragAB a[2], b[2];
#pragma unroll
        for (int i = 0; i < 2; ++i) {
            const bf16_t* ap = aBuf + (wm * 32 + i * 16 + l16) * LDP;
            a[i].q[0] = *(const uint4*)(ap + 8 * half);
            a[i].q[1] = *(const uint4*)(ap + 16 + 8 * half);
            const bf16_t* bp = bBuf + (wn * 32 + i * 16 + l16) * LDP;
            b[i].q[0] = *(const uint4*)(bp + 8 * half);
            b[i].q[1] = *(const uint4*)(bp + 16 + 8 * half);
        }

#pragma unroll
        for (int i = 0; i < 2; ++i)
#pragma unroll
            for (int j = 0; j < 2; ++j)
                acc[i][j] = __builtin_amdgcn_wmma_f32_16x16x32_bf16(
                    false, a[i].v, false, b[j].v, (short)0, acc[i][j],
                    false, false);

        buf ^= 1;
    }

    // epilogue: each wave writes its 32x32 tile
    const int mWave = mBase + wm * 32;
    const int nWave = nBase + wn * 32;
#pragma unroll
    for (int i = 0; i < 2; ++i) {
        const int mTop = mWave + i * 16 + 8 * half;
#pragma unroll
        for (int j = 0; j < 2; ++j) {
            const int nCol = nWave + j * 16 + l16;
            const float bv = bias ? bias[nCol] : 0.0f;
#pragma unroll
            for (int r = 0; r < 8; ++r) {
                const int m = mTop + r;
                float v = acc[i][j][r] * scale + bv;
                if (resid) v += resid[(size_t)m * ldo + nCol];
                if (outF)  outF[(size_t)m * ldo + nCol] = v;
                if (outH)  outH[(size_t)m * ldo + nCol] = f2bf(v);
            }
        }
    }
}

// ---------------------------------------------------------------------------
// 4) Per-row exact top-k threshold (4-pass 256-bin radix select on the
//    order-preserving u32 map) + masked softmax -> bf16 probabilities.
//    One block (256 thr) per row of 4096 scores (rows live in L2).
// ---------------------------------------------------------------------------
__global__ void __launch_bounds__(256)
k_topk_softmax(const float* __restrict__ scores, bf16_t* __restrict__ P,
               int kth) {
    const int S = SEQ;
    const int row = blockIdx.x;
    const int tid = threadIdx.x;
    const float* sr = scores + (size_t)row * S;

    __shared__ unsigned int hist[256];
    __shared__ unsigned int shPrefix, shRem;
    __shared__ float red[256];

    if (tid == 0) { shPrefix = 0u; shRem = (unsigned)kth; }
    __syncthreads();

    for (int pass = 0; pass < 4; ++pass) {
        const int shift = 24 - 8 * pass;
        hist[tid] = 0u;
        __syncthreads();
        const unsigned pre = shPrefix;
        for (int j = tid; j < S; j += 256) {
            const unsigned u = mapf(sr[j]);
            if (pass == 0 || (u >> (shift + 8)) == pre)
                atomicAdd(&hist[(u >> shift) & 255u], 1u);
        }
        __syncthreads();
        if (tid == 0) {
            unsigned rem = shRem, cum = 0;
            int bsel = 0;
            for (int b = 255; b >= 0; --b) {
                const unsigned h = hist[b];
                if (cum + h >= rem) { bsel = b; break; }
                cum += h;
            }
            shPrefix = (pre << 8) | (unsigned)bsel;
            shRem = rem - cum;
        }
        __syncthreads();
    }
    const unsigned thr = shPrefix;   // u32-mapped value of k-th largest score

    // row max (top-k always contains the max)
    float mx = -3.4e38f;
    for (int j = tid; j < S; j += 256) mx = fmaxf(mx, sr[j]);
    red[tid] = mx;
    __syncthreads();
    for (int o = 128; o > 0; o >>= 1) {
        if (tid < o) red[tid] = fmaxf(red[tid], red[tid + o]);
        __syncthreads();
    }
    mx = red[0];
    __syncthreads();

    // sum of exp over accepted elements
    float sum = 0.0f;
    for (int j = tid; j < S; j += 256) {
        const float s = sr[j];
        if (mapf(s) >= thr) sum += __expf(s - mx);
    }
    red[tid] = sum;
    __syncthreads();
    for (int o = 128; o > 0; o >>= 1) {
        if (tid < o) red[tid] += red[tid + o];
        __syncthreads();
    }
    const float inv = 1.0f / red[0];

    bf16_t* pr = P + (size_t)row * S;
    for (int j = tid; j < S; j += 256) {
        const float s = sr[j];
        const float p = (mapf(s) >= thr) ? __expf(s - mx) * inv : 0.0f;
        pr[j] = f2bf(p);
    }
}

// ---------------------------------------------------------------------------
// Launch
// ---------------------------------------------------------------------------
extern "C" void kernel_launch(void* const* d_in, const int* in_sizes, int n_in,
                              void* d_out, int out_size, void* d_ws, size_t ws_size,
                              hipStream_t stream) {
    (void)in_sizes; (void)n_in; (void)out_size; (void)ws_size;

    const float* x     = (const float*)d_in[0];
    const float* Wqkv  = (const float*)d_in[1];
    const float* bqkv  = (const float*)d_in[2];
    const float* Wout  = (const float*)d_in[3];
    const float* bout  = (const float*)d_in[4];
    const float* gamma = (const float*)d_in[5];
    const float* beta  = (const float*)d_in[6];
    float* out = (float*)d_out;

    // workspace carve-out (256B aligned slices)
    char* ws = (char*)d_ws;
    size_t off = 0;
    auto carve = [&](size_t bytes) -> void* {
        void* p = ws + off;
        off = (off + bytes + 255) & ~(size_t)255;
        return p;
    };
    float*  xnF     = (float*)  carve((size_t)MROWS * D_MODEL * 4);     // 32 MB
    bf16_t* xnH     = (bf16_t*) carve((size_t)MROWS * D_MODEL * 2);     // 16 MB
    bf16_t* WqkvT   = (bf16_t*) carve((size_t)N_QKV * D_MODEL * 2);     // 1.5 MB
    bf16_t* WoutT   = (bf16_t*) carve((size_t)D_MODEL * D_MODEL * 2);   // 0.5 MB
    bf16_t* qkvH    = (bf16_t*) carve((size_t)MROWS * N_QKV * 2);       // 48 MB
    bf16_t* vT      = (bf16_t*) carve((size_t)D_MODEL * SEQ * 2);       // 4 MB (reused/batch)
    float*  scoresF = (float*)  carve((size_t)SEQ * SEQ * 4);           // 64 MB (reused/batch)
    bf16_t* probH   = (bf16_t*) carve((size_t)SEQ * SEQ * 2);           // 32 MB (reused/batch)
    bf16_t* attnH   = (bf16_t*) carve((size_t)MROWS * D_MODEL * 2);     // 16 MB

    const float inv_sqrt_d = 0.044194173824159216f;  // 1/sqrt(512)

    // 1) LayerNorm
    k_layernorm<<<MROWS, 256, 0, stream>>>(x, gamma, beta, xnF, xnH);

    // 2) Weight transpose+convert to bf16 [N,K]
    k_transpose_cvt<<<dim3(N_QKV / 32, D_MODEL / 32), dim3(32, 8), 0, stream>>>(
        Wqkv, WqkvT, D_MODEL, N_QKV);
    k_transpose_cvt<<<dim3(D_MODEL / 32, D_MODEL / 32), dim3(32, 8), 0, stream>>>(
        Wout, WoutT, D_MODEL, D_MODEL);

    // 3) QKV GEMM: [16384,512] x [1536,512]^T -> bf16 [16384,1536]
    k_gemm_nt<<<dim3(N_QKV / BN, MROWS / BM), 256, 0, stream>>>(
        xnH, D_MODEL, WqkvT, D_MODEL, MROWS, N_QKV, D_MODEL,
        1.0f, bqkv, nullptr, nullptr, qkvH, N_QKV);

    // 4) Per-batch attention (score/prob buffers stay L2-resident)
    for (int b = 0; b < BATCH; ++b) {
        const bf16_t* qb = qkvH + (size_t)b * SEQ * N_QKV;            // q at col 0
        const bf16_t* kb = qb + D_MODEL;                              // k at col 512
        const bf16_t* vb = qb + 2 * D_MODEL;                          // v at col 1024

        // scores = q k^T / sqrt(D): [4096,512] x [4096,512]^T -> f32 [4096,4096]
        k_gemm_nt<<<dim3(SEQ / BN, SEQ / BM), 256, 0, stream>>>(
            qb, N_QKV, kb, N_QKV, SEQ, SEQ, D_MODEL,
            inv_sqrt_d, nullptr, nullptr, scoresF, nullptr, SEQ);

        // exact top-k threshold + masked softmax -> bf16 probabilities
        k_topk_softmax<<<SEQ, 256, 0, stream>>>(scoresF, probH, TOPK);

        // v^T for NT-form attn@v
        k_transpose_bf16<<<dim3(D_MODEL / 32, SEQ / 32), dim3(32, 8), 0, stream>>>(
            vb, N_QKV, vT, SEQ, D_MODEL);

        // attn_out = P v: [4096,4096] x [512,4096]^T -> bf16 [4096,512]
        k_gemm_nt<<<dim3(D_MODEL / BN, SEQ / BM), 256, 0, stream>>>(
            probH, SEQ, vT, SEQ, SEQ, D_MODEL, SEQ,
            1.0f, nullptr, nullptr, nullptr,
            attnH + (size_t)b * SEQ * D_MODEL, D_MODEL);
    }

    // 5) Output projection + bias + residual(xn): -> f32 d_out
    k_gemm_nt<<<dim3(D_MODEL / BN, MROWS / BM), 256, 0, stream>>>(
        attnH, D_MODEL, WoutT, D_MODEL, MROWS, D_MODEL, D_MODEL,
        1.0f, bout, xnF, out, nullptr, D_MODEL);
}
